// Attention_59270548684918
// MI455X (gfx1250) — matrix operand order repaired and checked by reading, not compiled, not run on hardware
//
#include <hip/hip_runtime.h>
#include <stdint.h>

// ---------------------------------------------------------------------------
// Problem constants (from reference): b=8, dim=256, n=2048, heads=8, d=64
// ---------------------------------------------------------------------------
constexpr int B_SZ   = 8;
constexpr int C_IN   = 256;
constexpr int N_SEQ  = 2048;
constexpr int HEADS  = 8;
constexpr int DHEAD  = 64;
constexpr int HID    = HEADS * DHEAD;      // 512
constexpr int QKVR   = 3 * HID;            // 1536
constexpr float SCALE = 0.125f;            // 64^-0.5

// GEMM tiling
constexpr int BDIM = 256;                  // 8 waves (wave32)
constexpr int BM   = 128;
constexpr int BN   = 64;
constexpr int KS   = 128;                  // K-stage depth in LDS
constexpr int KP   = 144;                  // padded LDS row stride (halves); 288B, 32B-aligned

typedef __attribute__((ext_vector_type(16))) __bf16 v16bf;
typedef __attribute__((ext_vector_type(8)))  __bf16 v8bf;
typedef __attribute__((ext_vector_type(8)))  float  v8f;
typedef __attribute__((ext_vector_type(4)))  float  v4f;
typedef __attribute__((ext_vector_type(4)))  unsigned int v4u;
typedef __attribute__((ext_vector_type(8)))  int    v8i;
typedef __attribute__((ext_vector_type(4)))  int    v4i;

// --- TDM availability / arity dispatch -------------------------------------
#if defined(__HIP_DEVICE_COMPILE__) && __has_builtin(__builtin_amdgcn_tensor_load_to_lds)
  #define USE_TDM 1
  #if __has_include(<hip/amd_detail/amd_gfx1250_TDM.h>)
    #define TDM_ARITY6 1
  #else
    #define TDM_ARITY6 0
  #endif
#else
  #define USE_TDM 0
#endif

__device__ __forceinline__ v8f wmma_bf16(v16bf a, v16bf b, v8f c) {
    return __builtin_amdgcn_wmma_f32_16x16x32_bf16(
        /*neg_a=*/false, a, /*neg_b=*/false, b,
        /*c_mod=*/(short)0, c, /*reuse_a=*/false, /*reuse_b=*/false);
}

// A fragment (16x32, MxK), row-major storage [row][stride kp]:
//   lanes 0-15 hold row m, K = 0-7 & 16-23 ; lanes 16-31 hold row m, K = 8-15 & 24-31
__device__ __forceinline__ v16bf load_A(const __bf16* base, int kp, int lane,
                                        int m0, int kk) {
    int row = m0 + (lane & 15);
    int ko  = kk + ((lane >> 4) << 3);
    const __bf16* p = base + row * kp + ko;
    union { v16bf v; v8bf h[2]; } u;
    u.h[0] = *(const v8bf*)(p);
    u.h[1] = *(const v8bf*)(p + 16);
    return u.v;
}

// B fragment (32x16, KxN), storage transposed [col][stride kp] (K contiguous):
//   lane j holds column j%16; lanes 0-15 K = 0-7 & 16-23 ; lanes 16-31 K = 8-15 & 24-31
__device__ __forceinline__ v16bf load_B(const __bf16* base, int kp, int lane,
                                        int n0, int kk) {
    int col = n0 + (lane & 15);
    int ko  = kk + ((lane >> 4) << 3);
    const __bf16* p = base + col * kp + ko;
    union { v16bf v; v8bf h[2]; } u;
    u.h[0] = *(const v8bf*)(p);
    u.h[1] = *(const v8bf*)(p + 16);
    return u.v;
}

#if USE_TDM
__device__ __forceinline__ void wait_tensorcnt0() {
#if __has_builtin(__builtin_amdgcn_s_wait_tensorcnt)
    __builtin_amdgcn_s_wait_tensorcnt(0);
#else
    asm volatile("s_wait_tensorcnt 0x0" ::: "memory");
#endif
}

// Issue a 2D bf16 tile load global->LDS via the Tensor Data Mover.
// Tile is tile_d0 (contiguous, elements) x tile_d1 rows; row stride = stride0.
// LDS destination is tile-contiguous at lds_off.
__device__ __forceinline__ void tdm_load_2d(const __bf16* gaddr, unsigned lds_off,
                                            unsigned tensor_d0, unsigned tensor_d1,
                                            unsigned stride0,
                                            unsigned tile_d0, unsigned tile_d1) {
    unsigned long long ga = (unsigned long long)(uintptr_t)gaddr;
    v4u g0;
    g0[0] = 1u;                                        // count=1 (valid), no gather
    g0[1] = lds_off;                                   // LDS byte address
    g0[2] = (unsigned)(ga & 0xFFFFFFFFu);              // global_addr[31:0]
    g0[3] = (unsigned)((ga >> 32) & 0x01FFFFFFu)       // global_addr[56:32]
            | (2u << 30);                              // type = 2 ("image")
    v8i g1;
    g1[0] = (int)(1u << 16);                           // data_size=1 -> 2 bytes/elem
    g1[1] = (int)((tensor_d0 & 0xFFFFu) << 16);        // tensor_dim0[15:0] (no atomic barrier)
    g1[2] = (int)(((tensor_d0 >> 16) & 0xFFFFu)        // tensor_dim0[31:16]
            | ((tensor_d1 & 0xFFFFu) << 16));          // tensor_dim1[15:0]
    g1[3] = (int)(((tensor_d1 >> 16) & 0xFFFFu)        // tensor_dim1[31:16]
            | ((tile_d0 & 0xFFFFu) << 16));            // tile_dim0
    g1[4] = (int)(tile_d1 & 0xFFFFu);                  // tile_dim1 (tile_dim2=0)
    g1[5] = (int)stride0;                              // tensor_dim0_stride[31:0]
    g1[6] = 0;                                         // stride0 hi / stride1 lo
    g1[7] = 0;
    v4i z4 = {0, 0, 0, 0};
#if TDM_ARITY6
    v8i z8 = {0, 0, 0, 0, 0, 0, 0, 0};
    __builtin_amdgcn_tensor_load_to_lds(g0, g1, z4, z4, z8, 0);
#else
    __builtin_amdgcn_tensor_load_to_lds(g0, g1, z4, z4, 0);
#endif
}

__device__ __forceinline__ unsigned lds_off_of(const void* p) {
    return (unsigned)(uintptr_t)p;   // LDS aperture: low 32 bits = LDS byte address
}
#endif  // USE_TDM

// ---------------------------------------------------------------------------
// Kernel 1: QKV projection. C(1536x2048) = W(1536x256) x X(256x2048) per batch.
// Epilogue: q,k -> [b][h][n][d] (bf16, d-contig), v -> [b][h][d][n] (n-contig).
// ---------------------------------------------------------------------------
__global__ __launch_bounds__(BDIM) void qkv_kernel(
    const float* __restrict__ x, const float* __restrict__ w,
    __bf16* __restrict__ q, __bf16* __restrict__ k, __bf16* __restrict__ v) {
    __shared__ __bf16 As[BM * KP];
    __shared__ __bf16 Bs[BN * KP];

    const int n0    = blockIdx.x * BN;
    const int m0g   = blockIdx.y * BM;
    const int batch = blockIdx.z;
    const int tid  = threadIdx.x;
    const int lane = tid & 31;
    const int wave = tid >> 5;
    const int wm = wave & 3, wn = wave >> 2;   // 4x2 wave grid, 32x32 per wave

    v8f acc[2][2] = {};
    for (int ks0 = 0; ks0 < C_IN; ks0 += KS) {
        __syncthreads();
        // Stage W tile [BM][KS] f32->bf16, row-major
        #pragma unroll
        for (int it = 0; it < (BM * KS / 4) / BDIM; ++it) {
            int idx = tid + it * BDIM;
            int row = idx / (KS / 4);
            int c4  = (idx % (KS / 4)) * 4;
            v4f wv = *(const v4f*)(w + (size_t)(m0g + row) * C_IN + ks0 + c4);
            __bf16* d = &As[row * KP + c4];
            d[0] = (__bf16)wv.x; d[1] = (__bf16)wv.y;
            d[2] = (__bf16)wv.z; d[3] = (__bf16)wv.w;
        }
        // Stage X tile transposed: Bs[j][c] = x[b][ks0+c][n0+j]  (f32->bf16)
        #pragma unroll 4
        for (int it = 0; it < (KS * BN) / BDIM; ++it) {
            int idx = tid + it * BDIM;
            int c = idx / BN;
            int j = idx % BN;
            Bs[j * KP + c] = (__bf16)x[((size_t)batch * C_IN + ks0 + c) * N_SEQ + n0 + j];
        }
        __syncthreads();
        #pragma unroll
        for (int kk = 0; kk < KS; kk += 32) {
            v16bf a0 = load_A(As, KP, lane, wm * 32,      kk);
            v16bf a1 = load_A(As, KP, lane, wm * 32 + 16, kk);
            v16bf b0 = load_B(Bs, KP, lane, wn * 32,      kk);
            v16bf b1 = load_B(Bs, KP, lane, wn * 32 + 16, kk);
            acc[0][0] = wmma_bf16(a0, b0, acc[0][0]);
            acc[0][1] = wmma_bf16(a0, b1, acc[0][1]);
            acc[1][0] = wmma_bf16(a1, b0, acc[1][0]);
            acc[1][1] = wmma_bf16(a1, b1, acc[1][1]);
        }
    }
    // Epilogue: C layout is lane=column(N), VGPR r = row M (+8 for upper lane half)
    #pragma unroll
    for (int mt = 0; mt < 2; ++mt)
    #pragma unroll
    for (int nt = 0; nt < 2; ++nt) {
        int nidx  = n0 + wn * 32 + nt * 16 + (lane & 15);
        int obase = m0g + wm * 32 + mt * 16 + ((lane >> 4) << 3);
        #pragma unroll
        for (int r = 0; r < 8; ++r) {
            int o = obase + r;
            float val = acc[mt][nt][r];
            if (o < HID) {                         // Q (scaled)
                int h = o >> 6, d = o & 63;
                q[(((size_t)batch * HEADS + h) * N_SEQ + nidx) * DHEAD + d] =
                    (__bf16)(val * SCALE);
            } else if (o < 2 * HID) {              // K
                int oo = o - HID, h = oo >> 6, d = oo & 63;
                k[(((size_t)batch * HEADS + h) * N_SEQ + nidx) * DHEAD + d] = (__bf16)val;
            } else {                               // V (d-major, n contiguous)
                int oo = o - 2 * HID, h = oo >> 6, d = oo & 63;
                v[(((size_t)batch * HEADS + h) * DHEAD + d) * N_SEQ + nidx] = (__bf16)val;
            }
        }
    }
}

// ---------------------------------------------------------------------------
// Kernel 2: flash attention. One block per (b,h, 128-query tile); 16 q/wave.
// K/V staged in 64-kj chunks; TDM double-buffered prefetch when available.
// ---------------------------------------------------------------------------
constexpr int CHK = 64;                    // kj chunk width
constexpr int NCH = N_SEQ / CHK;           // 32 chunks

__global__ __launch_bounds__(BDIM) void attn_kernel(
    const __bf16* __restrict__ q, const __bf16* __restrict__ k,
    const __bf16* __restrict__ v, __bf16* __restrict__ o) {
    __shared__ __bf16 ksh[2][CHK * DHEAD];    // [kj][d]
    __shared__ __bf16 vsh[2][DHEAD * CHK];    // [d][kj]
    __shared__ __bf16 psh[8 * 16 * 32];       // per-wave P transpose scratch

    const int tid = threadIdx.x, lane = tid & 31, wave = tid >> 5;
    const int bh    = blockIdx.y;             // batch*8 + head
    const int qbase = blockIdx.x * 128 + wave * 16;
    const __bf16* qp = q + (size_t)bh * N_SEQ * DHEAD;
    const __bf16* kp = k + (size_t)bh * N_SEQ * DHEAD;
    const __bf16* vp = v + (size_t)bh * DHEAD * N_SEQ;

    // Q fragments for this wave's 16 rows: 2 K-steps over d (reused all iters)
    v16bf qa[2];
    {
        const __bf16* p = qp + (size_t)(qbase + (lane & 15)) * DHEAD + ((lane >> 4) << 3);
        #pragma unroll
        for (int t = 0; t < 2; ++t) {
            union { v16bf v; v8bf h[2]; } u;
            u.h[0] = *(const v8bf*)(p + t * 32);
            u.h[1] = *(const v8bf*)(p + t * 32 + 16);
            qa[t] = u.v;
        }
    }

    v8f o_acc[4] = {};                        // 16 q x 64 d, fp32
    float m_run[8], l_run[8];
    #pragma unroll
    for (int r = 0; r < 8; ++r) { m_run[r] = -1e30f; l_run[r] = 0.f; }

#if USE_TDM
    // Prefetch chunk 0 via TDM (wave 0 only; EXEC is ignored by tensor ops,
    // so gate at wave granularity to issue exactly one DMA per descriptor).
    if (wave == 0) {
        tdm_load_2d(kp, lds_off_of(&ksh[0][0]), DHEAD, N_SEQ, DHEAD, DHEAD, CHK);
        tdm_load_2d(vp, lds_off_of(&vsh[0][0]), N_SEQ, DHEAD, N_SEQ, CHK, DHEAD);
        wait_tensorcnt0();
    }
    __syncthreads();
#endif

    for (int ch = 0; ch < NCH; ++ch) {
        const int kj0 = ch * CHK;
#if USE_TDM
        const int buf = ch & 1;
        // Issue DMA for next chunk into the other buffer (its last readers
        // finished before the barrier that ended the previous iteration).
        if (wave == 0 && ch + 1 < NCH) {
            const int nj = (ch + 1) * CHK;
            tdm_load_2d(kp + (size_t)nj * DHEAD, lds_off_of(&ksh[buf ^ 1][0]),
                        DHEAD, N_SEQ, DHEAD, DHEAD, CHK);
            tdm_load_2d(vp + nj, lds_off_of(&vsh[buf ^ 1][0]),
                        N_SEQ, DHEAD, N_SEQ, CHK, DHEAD);
        }
#else
        const int buf = 0;
        __syncthreads();
        #pragma unroll
        for (int it = 0; it < 2; ++it) {       // 2 x 256 sixteen-byte copies each
            int idx = tid + it * BDIM;         // 0..511
            int row = idx >> 3;                // 0..63
            int c8  = (idx & 7) * 8;
            *(v8bf*)&ksh[0][row * DHEAD + c8] =
                *(const v8bf*)(kp + (size_t)(kj0 + row) * DHEAD + c8);
            *(v8bf*)&vsh[0][row * CHK + c8] =
                *(const v8bf*)(vp + (size_t)row * N_SEQ + kj0 + c8);
        }
        __syncthreads();
#endif
        const __bf16* kbuf = &ksh[buf][0];
        const __bf16* vbuf = &vsh[buf][0];

        #pragma unroll
        for (int s = 0; s < 2; ++s) {          // two 32-kj substeps per chunk
            // S(16x32) = Q x K^T : B columns are kj, K dim is d
            v8f s0 = {}, s1 = {};
            #pragma unroll
            for (int t = 0; t < 2; ++t) {
                v16bf b0 = load_B(kbuf, DHEAD, lane, s * 32,      t * 32);
                v16bf b1 = load_B(kbuf, DHEAD, lane, s * 32 + 16, t * 32);
                s0 = wmma_bf16(qa[t], b0, s0);
                s1 = wmma_bf16(qa[t], b1, s1);
            }

            // Online softmax. Row data spread over 16-lane halves -> shfl_xor.
            float mnew[8], alpha[8];
            #pragma unroll
            for (int r = 0; r < 8; ++r) {
                float mx = fmaxf(s0[r], s1[r]);
                #pragma unroll
                for (int off = 8; off >= 1; off >>= 1)
                    mx = fmaxf(mx, __shfl_xor(mx, off, 32));
                mnew[r]  = fmaxf(m_run[r], mx);
                alpha[r] = __expf(m_run[r] - mnew[r]);
                m_run[r] = mnew[r];
            }
            #pragma unroll
            for (int r = 0; r < 8; ++r) {
                float p0 = __expf(s0[r] - mnew[r]);
                float p1 = __expf(s1[r] - mnew[r]);
                s0[r] = p0; s1[r] = p1;
                float rs = p0 + p1;
                #pragma unroll
                for (int off = 8; off >= 1; off >>= 1)
                    rs += __shfl_xor(rs, off, 32);
                l_run[r] = l_run[r] * alpha[r] + rs;
            }

            // Transpose P (C layout -> A layout) through per-wave LDS
            __bf16* pw = &psh[wave * 16 * 32];
            {
                int cA = lane & 15, cB = 16 + (lane & 15);
                int rb = (lane >> 4) << 3;
                #pragma unroll
                for (int r = 0; r < 8; ++r) {
                    pw[(rb + r) * 32 + cA] = (__bf16)s0[r];
                    pw[(rb + r) * 32 + cB] = (__bf16)s1[r];
                }
            }
            asm volatile("s_wait_dscnt 0" ::: "memory");  // same-wave LDS RAW
            v16bf pa;
            {
                const __bf16* p2 = pw + (lane & 15) * 32 + ((lane >> 4) << 3);
                union { v16bf v; v8bf h[2]; } u;
                u.h[0] = *(const v8bf*)(p2);
                u.h[1] = *(const v8bf*)(p2 + 16);
                pa = u.v;
            }

            // O = O*alpha + P(16x32) x V(32x64)
            #pragma unroll
            for (int t = 0; t < 4; ++t) {
                #pragma unroll
                for (int r = 0; r < 8; ++r) o_acc[t][r] *= alpha[r];
                v16bf bv = load_B(vbuf, CHK, lane, t * 16, s * 32);
                o_acc[t] = wmma_bf16(pa, bv, o_acc[t]);
            }
        }
#if USE_TDM
        if (wave == 0) wait_tensorcnt0();      // next chunk's DMA has landed
        __syncthreads();                       // all waves done with this buffer
#endif
    }

    // Normalize and store O as [b][i][h*64+d] (bf16) — B-operand layout for proj.
    {
        const int bb = bh >> 3, h = bh & 7;
        const int qh = qbase + ((lane >> 4) << 3);
        #pragma unroll
        for (int t = 0; t < 4; ++t) {
            int d = t * 16 + (lane & 15);
            #pragma unroll
            for (int r = 0; r < 8; ++r) {
                float val = o_acc[t][r] / l_run[r];
                o[((size_t)bb * N_SEQ + qh + r) * HID + h * DHEAD + d] = (__bf16)val;
            }
        }
    }
}

// ---------------------------------------------------------------------------
// Kernel 3: output projection. y(256x2048) = Wout(256x512) x O^T + bias.
// ---------------------------------------------------------------------------
__global__ __launch_bounds__(BDIM) void out_kernel(
    const __bf16* __restrict__ ow, const float* __restrict__ wout,
    const float* __restrict__ bout, float* __restrict__ y) {
    __shared__ __bf16 As[BM * KP];
    __shared__ __bf16 Bs[BN * KP];

    const int n0    = blockIdx.x * BN;
    const int m0g   = blockIdx.y * BM;
    const int batch = blockIdx.z;
    const int tid  = threadIdx.x;
    const int lane = tid & 31;
    const int wave = tid >> 5;
    const int wm = wave & 3, wn = wave >> 2;

    v8f acc[2][2] = {};
    for (int ks0 = 0; ks0 < HID; ks0 += KS) {
        __syncthreads();
        // Stage Wout tile [BM][KS] f32->bf16
        #pragma unroll
        for (int it = 0; it < (BM * KS / 4) / BDIM; ++it) {
            int idx = tid + it * BDIM;
            int row = idx / (KS / 4);
            int c4  = (idx % (KS / 4)) * 4;
            v4f wv = *(const v4f*)(wout + (size_t)(m0g + row) * HID + ks0 + c4);
            __bf16* d = &As[row * KP + c4];
            d[0] = (__bf16)wv.x; d[1] = (__bf16)wv.y;
            d[2] = (__bf16)wv.z; d[3] = (__bf16)wv.w;
        }
        // Stage O tile [BN][KS] bf16 (already [i][hd] layout: straight copy)
        #pragma unroll
        for (int it = 0; it < (BN * KS / 8) / BDIM; ++it) {
            int idx = tid + it * BDIM;
            int j  = idx / (KS / 8);
            int c8 = (idx % (KS / 8)) * 8;
            *(v8bf*)&Bs[j * KP + c8] =
                *(const v8bf*)(ow + ((size_t)batch * N_SEQ + n0 + j) * HID + ks0 + c8);
        }
        __syncthreads();
        #pragma unroll
        for (int kk = 0; kk < KS; kk += 32) {
            v16bf a0 = load_A(As, KP, lane, wm * 32,      kk);
            v16bf a1 = load_A(As, KP, lane, wm * 32 + 16, kk);
            v16bf b0 = load_B(Bs, KP, lane, wn * 32,      kk);
            v16bf b1 = load_B(Bs, KP, lane, wn * 32 + 16, kk);
            acc[0][0] = wmma_bf16(a0, b0, acc[0][0]);
            acc[0][1] = wmma_bf16(a0, b1, acc[0][1]);
            acc[1][0] = wmma_bf16(a1, b0, acc[1][0]);
            acc[1][1] = wmma_bf16(a1, b1, acc[1][1]);
        }
    }
    #pragma unroll
    for (int mt = 0; mt < 2; ++mt)
    #pragma unroll
    for (int nt = 0; nt < 2; ++nt) {
        int nidx  = n0 + wn * 32 + nt * 16 + (lane & 15);
        int obase = m0g + wm * 32 + mt * 16 + ((lane >> 4) << 3);
        #pragma unroll
        for (int r = 0; r < 8; ++r) {
            int oo = obase + r;
            y[((size_t)batch * C_IN + oo) * N_SEQ + nidx] = acc[mt][nt][r] + bout[oo];
        }
    }
}

// ---------------------------------------------------------------------------
extern "C" void kernel_launch(void* const* d_in, const int* in_sizes, int n_in,
                              void* d_out, int out_size, void* d_ws, size_t ws_size,
                              hipStream_t stream) {
    const float* x     = (const float*)d_in[0];
    const float* w_qkv = (const float*)d_in[1];
    const float* w_out = (const float*)d_in[2];
    const float* b_out = (const float*)d_in[3];
    float* y = (float*)d_out;

    const size_t seg = (size_t)B_SZ * HEADS * N_SEQ * DHEAD;   // 8.4M elems
    __bf16* q = (__bf16*)d_ws;
    __bf16* k = q + seg;
    __bf16* v = k + seg;
    __bf16* o = v + seg;          // [b][n][512] bf16

    dim3 blk(BDIM);
    dim3 g1(N_SEQ / BN, QKVR / BM, B_SZ);      // 32 x 12 x 8
    qkv_kernel<<<g1, blk, 0, stream>>>(x, w_qkv, q, k, v);

    dim3 g2(N_SEQ / 128, B_SZ * HEADS);        // 16 x 64
    attn_kernel<<<g2, blk, 0, stream>>>(q, k, v, o);

    dim3 g3(N_SEQ / BN, C_IN / BM, B_SZ);      // 32 x 2 x 8
    out_kernel<<<g3, blk, 0, stream>>>(o, w_out, b_out, y);
}
